// TextLSTM_56796647522395
// MI455X (gfx1250) — matrix operand order repaired and checked by reading, not compiled, not run on hardware
//
#include <hip/hip_runtime.h>
#include <hip/hip_bf16.h>

// ---------- problem constants ----------
#define N_CLASS 32000
#define EMB     512
#define HID     1024
#define GATES   (4*HID)        // 4096
#define BB      64             // batch
#define TT      256            // seq len
#define KTOT    (EMB + HID)    // 1536 fused K (x-proj + h-proj)
#define KCH     (KTOT/32)      // 48 K-chunks of 32
#define KCH_X   (EMB/32)       // 16 chunks from embeddings
#define CTILES  (GATES/16)     // 256 column tiles of 16

typedef __attribute__((ext_vector_type(16))) __bf16    bf16x16;
typedef __attribute__((ext_vector_type(2)))  __bf16    bf16x2;
typedef __attribute__((ext_vector_type(8)))  float     v8f;
typedef __attribute__((ext_vector_type(8)))  unsigned  u32x8;

static __device__ __forceinline__ unsigned pack_bf16(float a, float b) {
  bf16x2 p; p[0] = (__bf16)a; p[1] = (__bf16)b;
  return __builtin_bit_cast(unsigned, p);
}
static __device__ __forceinline__ float sigmoidf_(float x) {
  return 1.0f / (1.0f + __expf(-x));
}
// A-fragment for row-major 16-bit A: two contiguous 16B runs per lane.
static __device__ __forceinline__ bf16x16 load_a_frag(const __bf16* rowp,
                                                      unsigned elemoff) {
  uint4 a0 = *(const uint4*)(rowp + elemoff);        // K 2j+8*khalf, j<4
  uint4 a1 = *(const uint4*)(rowp + elemoff + 16);   // K 16+2j+8*khalf
  u32x8 araw;
  araw[0] = a0.x; araw[1] = a0.y; araw[2] = a0.z; araw[3] = a0.w;
  araw[4] = a1.x; araw[5] = a1.y; araw[6] = a1.z; araw[7] = a1.w;
  return __builtin_bit_cast(bf16x16, araw);
}

// ---------------------------------------------------------------------------
// Pack [W_x ; W_h] (fp32, K rows x 4096 cols) into bf16 WMMA B-operand
// fragment order: packedW[((ct*KCH + kc)*32 + lane)*8 + j] holds the bf16
// pair for lane (n = lane&15, khalf = lane>>4), K = kc*32 + 16*(j>>2) + khalf*8 + 2*(j&3).
// ---------------------------------------------------------------------------
__global__ __launch_bounds__(256) void pack_w_kernel(
    const float* __restrict__ Wx, const float* __restrict__ Wh,
    unsigned* __restrict__ packedW) {
  unsigned idx  = blockIdx.x * 256u + threadIdx.x;     // one u32
  unsigned j    = idx & 7u;
  unsigned lane = (idx >> 3) & 31u;
  unsigned rest = idx >> 8;
  unsigned kc   = rest % KCH;
  unsigned ct   = rest / KCH;
  unsigned n = lane & 15u, khalf = lane >> 4;
  unsigned k   = kc * 32u + (j >> 2) * 16u + khalf * 8u + 2u * (j & 3u);      // B-fragment K order
  unsigned col = ct * 16u + n;
  float x0 = (k     < EMB) ? Wx[(size_t)k * GATES + col]
                           : Wh[(size_t)(k - EMB) * GATES + col];
  float x1 = (k + 1 < EMB) ? Wx[(size_t)(k + 1) * GATES + col]
                           : Wh[(size_t)(k + 1 - EMB) * GATES + col];
  packedW[idx] = pack_bf16(x0, x1);
}

// ---------------------------------------------------------------------------
// Embedding gather + fp32 -> bf16: XE[b*T + t][e] row-major bf16.
// ---------------------------------------------------------------------------
__global__ __launch_bounds__(256) void gather_emb_kernel(
    const int* __restrict__ X, const float* __restrict__ emb,
    unsigned* __restrict__ XEu) {
  unsigned i   = blockIdx.x * 256u + threadIdx.x;      // one u32 (2 elems)
  unsigned e   = (i * 2u) % EMB;
  unsigned row = (i * 2u) / EMB;                       // b*T + t
  int tok = X[row];
  const float* s = emb + (size_t)tok * EMB + e;
  XEu[i] = pack_bf16(s[0], s[1]);
}

// ---------------------------------------------------------------------------
// One LSTM timestep. One WAVE per block: wave owns a (16-row x 16-hidden)
// patch and all 4 gates for it. g = b + [xe_t | H] @ [W_x ; W_h] via bf16
// WMMA; the four gate accumulators share the (VGPR,lane)->(row,col) map, so
// the sigmoid/tanh/C/H update is lane-local. Zero LDS, zero barriers.
// Grid: 256 = 64 hidden stripes x 4 row tiles. H double-buffered.
// ---------------------------------------------------------------------------
__global__ __launch_bounds__(32) void lstm_step_kernel(
    const unsigned* __restrict__ packedW,
    const __bf16*  __restrict__ XE,    // [B*T][EMB]
    const float*   __restrict__ bias,  // [4H]
    const __bf16*  __restrict__ Hin,   // [B][HID]
    __bf16*        __restrict__ Hout,  // [B][HID]
    float*         __restrict__ C,     // [B][HID]
    int t) {
  const unsigned lane   = threadIdx.x & 31u;
  const unsigned stripe = blockIdx.x & 63u;   // hidden stripe (16 cols)
  const unsigned r      = blockIdx.x >> 6;    // row tile (16 rows)
  const unsigned n = lane & 15u, khalf = lane >> 4;
  const unsigned hb = stripe * 16u;
  const unsigned myrow = r * 16u + n;         // A-matrix row for this lane

  // accumulators for gates i,f,o,c ; bias splat (bias depends on col only)
  v8f acc[4];
  #pragma unroll
  for (int g = 0; g < 4; ++g) {
    float bv = bias[(size_t)g * HID + hb + n];
    #pragma unroll
    for (int v = 0; v < 8; ++v) acc[g][v] = bv;
  }

  const u32x8* PB = (const u32x8*)packedW;
  unsigned bbase[4];
  #pragma unroll
  for (int g = 0; g < 4; ++g)
    bbase[g] = ((unsigned)g * 64u + stripe) * (KCH * 32u) + lane;

  const __bf16* arow_x = XE + ((size_t)myrow * TT + t) * EMB;
  const __bf16* arow_h = Hin + (size_t)myrow * HID;
  const unsigned kh8 = khalf * 8u;

  // K over embedding part (chunks 0..15)
  for (int kc = 0; kc < KCH_X; ++kc) {
    bf16x16 af = load_a_frag(arow_x, (unsigned)kc * 32u + kh8);
    #pragma unroll
    for (int g = 0; g < 4; ++g) {
      bf16x16 bf = __builtin_bit_cast(bf16x16, PB[bbase[g] + (unsigned)kc * 32u]);
      acc[g] = __builtin_amdgcn_wmma_f32_16x16x32_bf16(false, af, false, bf,
                                                       (short)0, acc[g], false, false);
    }
  }
  // K over recurrent part (chunks 16..47)
  for (int kc = KCH_X; kc < KCH; ++kc) {
    bf16x16 af = load_a_frag(arow_h, (unsigned)(kc - KCH_X) * 32u + kh8);
    #pragma unroll
    for (int g = 0; g < 4; ++g) {
      bf16x16 bf = __builtin_bit_cast(bf16x16, PB[bbase[g] + (unsigned)kc * 32u]);
      acc[g] = __builtin_amdgcn_wmma_f32_16x16x32_bf16(false, af, false, bf,
                                                       (short)0, acc[g], false, false);
    }
  }

  // lane-local LSTM cell update: 8 elements per lane (C/D tile layout)
  #pragma unroll
  for (int v = 0; v < 8; ++v) {
    unsigned row = r * 16u + (khalf ? (unsigned)v + 8u : (unsigned)v);
    size_t ci = (size_t)row * HID + hb + n;
    float cNew = sigmoidf_(acc[1][v]) * C[ci]
               + sigmoidf_(acc[0][v]) * tanhf(acc[3][v]);
    C[ci] = cNew;
    Hout[ci] = (__bf16)(sigmoidf_(acc[2][v]) * tanhf(cNew));
  }
}

// ---------------------------------------------------------------------------
// Logits: out[64,32000] = H @ W_hq + b_q. W_hq streams from HBM exactly ONCE:
// each chunk's 32x32 fp32 block is cooperatively converted to bf16 fragment
// order in LDS, then all 8 waves (4 row tiles x 2 col tiles) consume it.
// A fragments come straight from row-major bf16 H (two b128 per lane).
// ---------------------------------------------------------------------------
__global__ __launch_bounds__(256) void logits_kernel(
    const __bf16* __restrict__ H,     // [B][HID] final state
    const float*  __restrict__ Whq,   // [HID][N_CLASS]
    const float*  __restrict__ bq,    // [N_CLASS]
    float*        __restrict__ out) { // [B][N_CLASS]
  __shared__ unsigned Bstage[512];    // 2 col tiles x 32 lanes x 8 u32
  const unsigned tid = threadIdx.x, lane = tid & 31u, wave = tid >> 5;
  const unsigned r = wave & 3u, cloc = wave >> 2;
  const unsigned n = lane & 15u, khalf = lane >> 4;
  const unsigned col = blockIdx.x * 32u + cloc * 16u + n;

  v8f acc;
  { float b = bq[col];
    #pragma unroll
    for (int v = 0; v < 8; ++v) acc[v] = b; }

  const __bf16* arow = H + (size_t)(r * 16u + n) * HID;
  const unsigned kh8 = khalf * 8u;

  for (int kc = 0; kc < HID / 32; ++kc) {
    // cooperative fp32 -> bf16 fragment conversion (512 u32, 2 per thread)
    #pragma unroll
    for (int s = 0; s < 2; ++s) {
      unsigned u  = (unsigned)s * 256u + tid;
      unsigned ct = u >> 8, l = (u >> 3) & 31u, j = u & 7u;
      unsigned nn = l & 15u, kh = l >> 4;
      unsigned k  = (unsigned)kc * 32u + (j >> 2) * 16u + kh * 8u + 2u * (j & 3u);
      unsigned cc = blockIdx.x * 32u + ct * 16u + nn;
      Bstage[u] = pack_bf16(Whq[(size_t)k * N_CLASS + cc],
                            Whq[(size_t)(k + 1) * N_CLASS + cc]);
    }
    __syncthreads();

    bf16x16 af = load_a_frag(arow, (unsigned)kc * 32u + kh8);
    u32x8 braw = *(const u32x8*)&Bstage[cloc * 256u + lane * 8u];
    bf16x16 bf = __builtin_bit_cast(bf16x16, braw);
    acc = __builtin_amdgcn_wmma_f32_16x16x32_bf16(false, af, false, bf,
                                                  (short)0, acc, false, false);
    __syncthreads();
  }

  #pragma unroll
  for (int v = 0; v < 8; ++v) {
    unsigned row = r * 16u + (khalf ? (unsigned)v + 8u : (unsigned)v);
    out[(size_t)row * N_CLASS + col] = acc[v];
  }
}

// ---------------------------------------------------------------------------
extern "C" void kernel_launch(void* const* d_in, const int* in_sizes, int n_in,
                              void* d_out, int out_size, void* d_ws, size_t ws_size,
                              hipStream_t stream) {
  (void)in_sizes; (void)n_in; (void)out_size; (void)ws_size;
  const int*   X    = (const int*)d_in[0];
  const float* emb  = (const float*)d_in[1];
  const float* Wx   = (const float*)d_in[2];
  const float* Wh   = (const float*)d_in[3];
  const float* bias = (const float*)d_in[4];
  const float* Whq  = (const float*)d_in[5];
  const float* bq   = (const float*)d_in[6];
  float* out = (float*)d_out;

  // workspace layout (~28.5 MB total)
  char* ws = (char*)d_ws;
  size_t o = 0;
  unsigned* packedW = (unsigned*)(ws + o); o += (size_t)CTILES * KCH * 32 * 8 * 4; // 12 MB
  __bf16*   XE      = (__bf16*)(ws + o);   o += (size_t)BB * TT * EMB * 2;         // 16.78 MB
  __bf16*   H0      = (__bf16*)(ws + o);   o += (size_t)BB * HID * 2;
  __bf16*   H1      = (__bf16*)(ws + o);   o += (size_t)BB * HID * 2;
  float*    C       = (float*)(ws + o);    o += (size_t)BB * HID * 4;

  // per-launch weight pack + embedding gather + state init
  pack_w_kernel<<<(CTILES * KCH * 32 * 8) / 256, 256, 0, stream>>>(Wx, Wh, packedW);
  gather_emb_kernel<<<((size_t)BB * TT * EMB / 2) / 256, 256, 0, stream>>>(X, emb, (unsigned*)XE);
  hipMemsetAsync(H0, 0, (size_t)BB * HID * 2, stream);
  hipMemsetAsync(C,  0, (size_t)BB * HID * 4, stream);

  // sequential recurrence: 256 step kernels, barrier-free single-wave blocks
  __bf16* Hin = H0; __bf16* Hout = H1;
  for (int t = 0; t < TT; ++t) {
    lstm_step_kernel<<<4 * (HID / 16), 32, 0, stream>>>(packedW, XE, bias, Hin, Hout, C, t);
    __bf16* tmp = Hin; Hin = Hout; Hout = tmp;
  }

  // final projection (HBM-bound, W_hq read exactly once)
  logits_kernel<<<N_CLASS / 32, 256, 0, stream>>>(Hin, Whq, bq, out);
}
